// TextRNN_59416577573439
// MI455X (gfx1250) — compile-verified, matrix operand order checked by reading, not run on hardware
//
#include <hip/hip_runtime.h>
#include <hip/hip_bf16.h>
#include <cstddef>

// ---------------------------------------------------------------- types
typedef __attribute__((ext_vector_type(16))) __bf16 v16bf;
typedef __attribute__((ext_vector_type(8)))  float  v8f;

union Frag16 { v16bf v; uint4 q[2]; };

#define N_CLASS 50257
#define N_PAD   50272   // 3142 * 16, multiple of 32
#define EMB     256
#define HID     512
#define BATCH   128
#define SEQ     256

// ---------------------------------------------------------------- helpers
__device__ __forceinline__ unsigned short f2bf(float f) {
    unsigned int u = __float_as_uint(f);
    if ((u & 0x7fffffffu) > 0x7f800000u) return (unsigned short)((u >> 16) | 0x40); // quiet NaN
    unsigned int r = u + 0x7fffu + ((u >> 16) & 1u); // round-to-nearest-even
    return (unsigned short)(r >> 16);
}

__device__ __forceinline__ float sigm(float x) { return 1.0f / (1.0f + __expf(-x)); }

__device__ __forceinline__ v8f splat8(float x) {
    v8f v = { x, x, x, x, x, x, x, x };
    return v;
}

// Load one 16x32 bf16 WMMA fragment given a precomputed per-lane pointer.
// Lane pointer already encodes: row (lane%16) * ld + (lane>=16 ? 8 : 0).
// ISA lane layout: lanes 0-15 hold K in [k0,k0+8) u [k0+16,k0+24),
//                  lanes 16-31 hold K in [k0+8,k0+16) u [k0+24,k0+32).
__device__ __forceinline__ v16bf load2(const unsigned short* __restrict__ p) {
    Frag16 f;
    f.q[0] = *(const uint4*)(p);
    f.q[1] = *(const uint4*)(p + 16);
    return f.v;
}

__device__ __forceinline__ v8f wmma_bf16(v16bf a, v16bf b, v8f c) {
    return __builtin_amdgcn_wmma_f32_16x16x32_bf16(false, a, false, b,
                                                   (short)0, c, false, false);
}

// ---------------------------------------------------------------- prep kernels

// Transpose + convert: src f32 [K][N] row-major -> dst bf16 [Npad][K] row-major.
// gridDim = (Npad/32, ceil(K/32)), block = (32, 8). Rows n >= N are zero-filled.
__global__ __launch_bounds__(256)
void k_transpose_bf16(const float* __restrict__ src, unsigned short* __restrict__ dst,
                      int K, int N) {
    __shared__ float t[32][33];
    int bx = blockIdx.x, by = blockIdx.y;
    int tx = threadIdx.x, ty = threadIdx.y;
#pragma unroll
    for (int j = 0; j < 4; ++j) {
        int k = by * 32 + ty + j * 8;
        int n = bx * 32 + tx;
        float v = 0.0f;
        if (k < K && n < N) v = src[(size_t)k * N + n];
        t[ty + j * 8][tx] = v;
    }
    __syncthreads();
#pragma unroll
    for (int j = 0; j < 4; ++j) {
        int n = bx * 32 + ty + j * 8;
        int k = by * 32 + tx;
        if (k < K) dst[(size_t)n * K + k] = f2bf(t[tx][ty + j * 8]);
    }
}

// Embedding gather + convert, stored time-major: Xe[s][b][e] bf16.
// grid = SEQ*BATCH blocks (block = s*BATCH + b), 256 threads (e).
__global__ __launch_bounds__(256)
void k_embed(const int* __restrict__ X, const float* __restrict__ Cemb,
             unsigned short* __restrict__ Xe) {
    int blk = blockIdx.x;
    int s = blk >> 7;          // / BATCH
    int b = blk & 127;         // % BATCH
    int tok = X[(size_t)b * SEQ + s];
    int e = threadIdx.x;
    Xe[(size_t)blk * EMB + e] = f2bf(Cemb[(size_t)tok * EMB + e]);
}

__global__ __launch_bounds__(256)
void k_init_hc(unsigned short* __restrict__ H0, float* __restrict__ C) {
    int i = blockIdx.x * blockDim.x + threadIdx.x;
    if (i < BATCH * HID) { H0[i] = 0; C[i] = 0.0f; }
}

// ---------------------------------------------------------------- fused LSTM step
// gates[b, g, h] = Xe_s[b,:] @ WxT[g*HID+h,:]^T + Hin[b,:] @ WhT[g*HID+h,:]^T + b_g[h]
// Gate order g: 0=f, 1=i, 2=c~, 3=o.  Then C/H elementwise update, H written bf16.
// grid = HID/32 = 16 blocks (h-tiles of 32), block = 256 threads = 8 waves.
// wave w owns batch rows [16w, 16w+16); loops over 2 N-tiles of 16 within h-tile.
__global__ __launch_bounds__(256)
void k_lstm_step(const unsigned short* __restrict__ Xe_s,   // [BATCH][EMB]
                 const unsigned short* __restrict__ Hin,    // [BATCH][HID]
                 unsigned short* __restrict__ Hout,         // [BATCH][HID]
                 float* __restrict__ C,                     // [BATCH][HID]
                 const unsigned short* __restrict__ WxT,    // [4*HID][EMB]
                 const unsigned short* __restrict__ WhT,    // [4*HID][HID]
                 const float* __restrict__ bF, const float* __restrict__ bI,
                 const float* __restrict__ bC, const float* __restrict__ bO) {
    int lane  = threadIdx.x & 31;
    int mt    = threadIdx.x >> 5;        // wave id -> batch tile
    int hbase = blockIdx.x * 32;
    int lrow  = lane & 15;
    int c0    = (lane & 16) ? 8 : 0;     // K half-offset per ISA fragment layout

    // per-lane A fragment base pointers (constant across ht / kt)
    const unsigned short* pAx = Xe_s + (size_t)(mt * 16 + lrow) * EMB + c0;
    const unsigned short* pAh = Hin  + (size_t)(mt * 16 + lrow) * HID + c0;

    for (int ht = 0; ht < 2; ++ht) {
        int nb = hbase + ht * 16;
        int h  = nb + lrow;
        int rB = nb + lrow;

        // per-lane B fragment base pointers, one per gate
        const unsigned short* pBx0 = WxT + (size_t)(0 * HID + rB) * EMB + c0;
        const unsigned short* pBx1 = WxT + (size_t)(1 * HID + rB) * EMB + c0;
        const unsigned short* pBx2 = WxT + (size_t)(2 * HID + rB) * EMB + c0;
        const unsigned short* pBx3 = WxT + (size_t)(3 * HID + rB) * EMB + c0;
        const unsigned short* pBh0 = WhT + (size_t)(0 * HID + rB) * HID + c0;
        const unsigned short* pBh1 = WhT + (size_t)(1 * HID + rB) * HID + c0;
        const unsigned short* pBh2 = WhT + (size_t)(2 * HID + rB) * HID + c0;
        const unsigned short* pBh3 = WhT + (size_t)(3 * HID + rB) * HID + c0;

        // fold gate biases into accumulator init (bias depends only on h = lane col)
        float vbF = bF[h], vbI = bI[h], vbC = bC[h], vbO = bO[h];
        v8f aF = splat8(vbF), aI = splat8(vbI), aC = splat8(vbC), aO = splat8(vbO);

        // ---- K over x-contribution: 8 branch-free steps of 32
#pragma unroll 4
        for (int kt = 0; kt < EMB / 32; ++kt) {
            int off = kt * 32;
            v16bf a = load2(pAx + off);
            aF = wmma_bf16(a, load2(pBx0 + off), aF);
            aI = wmma_bf16(a, load2(pBx1 + off), aI);
            aC = wmma_bf16(a, load2(pBx2 + off), aC);
            aO = wmma_bf16(a, load2(pBx3 + off), aO);
        }
        // ---- K over h-contribution: 16 branch-free steps of 32
#pragma unroll 4
        for (int kt = 0; kt < HID / 32; ++kt) {
            int off = kt * 32;
            v16bf a = load2(pAh + off);
            aF = wmma_bf16(a, load2(pBh0 + off), aF);
            aI = wmma_bf16(a, load2(pBh1 + off), aI);
            aC = wmma_bf16(a, load2(pBh2 + off), aC);
            aO = wmma_bf16(a, load2(pBh3 + off), aO);
        }

        // elementwise LSTM update; D layout: m = 16*mt + r + 8*(lane>=16), n = h
        int mofs = mt * 16 + ((lane & 16) ? 8 : 0);
#pragma unroll
        for (int r = 0; r < 8; ++r) {
            int m = mofs + r;
            float f  = sigm(aF[r]);
            float i  = sigm(aI[r]);
            float ct = tanhf(aC[r]);
            float o  = sigm(aO[r]);
            size_t idx = (size_t)m * HID + h;
            float cNew = f * C[idx] + i * ct;
            C[idx]     = cNew;
            Hout[idx]  = f2bf(o * tanhf(cNew));
        }
    }
}

// ---------------------------------------------------------------- final projection
// out[b, n] = H[b,:] @ WqT[n,:]^T + b_q[n]; grid = N_PAD/16 = 3142, block = 256.
__global__ __launch_bounds__(256)
void k_proj(const unsigned short* __restrict__ Hfin,  // [BATCH][HID] bf16
            const unsigned short* __restrict__ WqT,   // [N_PAD][HID] bf16 (padded rows zero)
            const float* __restrict__ bq,
            float* __restrict__ out) {                // [BATCH][N_CLASS]
    int lane = threadIdx.x & 31;
    int mt   = threadIdx.x >> 5;
    int nb   = blockIdx.x * 16;
    int lrow = lane & 15;
    int c0   = (lane & 16) ? 8 : 0;
    int n    = nb + lrow;

    const unsigned short* pA = Hfin + (size_t)(mt * 16 + lrow) * HID + c0;
    const unsigned short* pB = WqT  + (size_t)(nb + lrow) * HID + c0;

    // fold bias into accumulator init; clamp read for the padded N tail
    float bias = bq[n < N_CLASS ? n : 0];
    v8f acc = splat8(bias);

#pragma unroll 8
    for (int kt = 0; kt < HID / 32; ++kt) {
        int off = kt * 32;
        acc = wmma_bf16(load2(pA + off), load2(pB + off), acc);
    }

    if (n < N_CLASS) {
        int mofs = mt * 16 + ((lane & 16) ? 8 : 0);
#pragma unroll
        for (int r = 0; r < 8; ++r) {
            int m = mofs + r;
            out[(size_t)m * N_CLASS + n] = acc[r];
        }
    }
}

// ---------------------------------------------------------------- host launcher
extern "C" void kernel_launch(void* const* d_in, const int* in_sizes, int n_in,
                              void* d_out, int out_size, void* d_ws, size_t ws_size,
                              hipStream_t stream) {
    const int*   X     = (const int*)  d_in[0];
    const float* Cemb  = (const float*)d_in[1];
    const float* W_xi  = (const float*)d_in[2];
    const float* W_xf  = (const float*)d_in[3];
    const float* W_xo  = (const float*)d_in[4];
    const float* W_xc  = (const float*)d_in[5];
    const float* W_hi  = (const float*)d_in[6];
    const float* W_hf  = (const float*)d_in[7];
    const float* W_ho  = (const float*)d_in[8];
    const float* W_hc  = (const float*)d_in[9];
    const float* b_i   = (const float*)d_in[10];
    const float* b_f   = (const float*)d_in[11];
    const float* b_o   = (const float*)d_in[12];
    const float* b_c   = (const float*)d_in[13];
    const float* W_hq  = (const float*)d_in[14];
    const float* b_q   = (const float*)d_in[15];

    // workspace carve-up (256-byte aligned)
    char* ws = (char*)d_ws;
    size_t off = 0;
    auto carve = [&](size_t bytes) {
        size_t o = off;
        off = (off + bytes + 255) & ~(size_t)255;
        return o;
    };
    unsigned short* WxT  = (unsigned short*)(ws + carve((size_t)4 * HID * EMB * 2)); // 1 MB
    unsigned short* WhT  = (unsigned short*)(ws + carve((size_t)4 * HID * HID * 2)); // 2 MB
    unsigned short* WqT  = (unsigned short*)(ws + carve((size_t)N_PAD * HID * 2));   // 51.5 MB
    unsigned short* Xe   = (unsigned short*)(ws + carve((size_t)SEQ * BATCH * EMB * 2)); // 16 MB
    unsigned short* H0   = (unsigned short*)(ws + carve((size_t)BATCH * HID * 2));
    unsigned short* H1   = (unsigned short*)(ws + carve((size_t)BATCH * HID * 2));
    float*          Cst  = (float*)         (ws + carve((size_t)BATCH * HID * 4));
    (void)ws_size; (void)in_sizes; (void)n_in; (void)out_size;

    dim3 tb(32, 8);
    // gate order f, i, c, o
    const float* wx[4] = { W_xf, W_xi, W_xc, W_xo };
    const float* wh[4] = { W_hf, W_hi, W_hc, W_ho };
    for (int g = 0; g < 4; ++g) {
        k_transpose_bf16<<<dim3(HID / 32, EMB / 32), tb, 0, stream>>>(
            wx[g], WxT + (size_t)g * HID * EMB, EMB, HID);
        k_transpose_bf16<<<dim3(HID / 32, HID / 32), tb, 0, stream>>>(
            wh[g], WhT + (size_t)g * HID * HID, HID, HID);
    }
    k_transpose_bf16<<<dim3(N_PAD / 32, HID / 32), tb, 0, stream>>>(W_hq, WqT, HID, N_CLASS);

    k_embed<<<SEQ * BATCH, 256, 0, stream>>>(X, Cemb, Xe);
    k_init_hc<<<(BATCH * HID + 255) / 256, 256, 0, stream>>>(H0, Cst);

    for (int s = 0; s < SEQ; ++s) {
        const unsigned short* Xe_s = Xe + (size_t)s * BATCH * EMB;
        const unsigned short* Hin  = (s & 1) ? H1 : H0;
        unsigned short*       Hout = (s & 1) ? H0 : H1;
        k_lstm_step<<<HID / 32, 256, 0, stream>>>(Xe_s, Hin, Hout, Cst, WxT, WhT,
                                                  b_f, b_i, b_c, b_o);
    }
    // SEQ = 256 (even): last step (s = 255, odd) wrote H0
    k_proj<<<N_PAD / 16, 256, 0, stream>>>(H0, WqT, b_q, (float*)d_out);
}
